// Seq2Seq_71107478553198
// MI455X (gfx1250) — matrix operand order repaired and checked by reading, not compiled, not run on hardware
//
#include <hip/hip_runtime.h>
#include <stdint.h>
#include <stddef.h>

// Problem constants (reference: T, N, C, D, K = 256, 4096, 12, 256, 12)
#define T_STEPS 256
#define NN      4096
#define CC      12
#define DD      256
#define KK      12

#define WAVES          2
#define ROWS_PER_WAVE  16
#define ROWS_PER_BLOCK (WAVES * ROWS_PER_WAVE)   // 32
#define NBLOCKS        (NN / ROWS_PER_BLOCK)     // 128
#define NTHREADS       (WAVES * 32)              // 64

typedef __attribute__((ext_vector_type(16))) __bf16        v16bf;
typedef __attribute__((ext_vector_type(8)))  float         v8f;
typedef __attribute__((ext_vector_type(8)))  unsigned int  v8u;

// accurate RNE f32->bf16 (setup-time fills only)
static __device__ __forceinline__ unsigned short f2bf_rne(float f) {
  unsigned int u = __float_as_uint(f);
  u += 0x7fffu + ((u >> 16) & 1u);
  return (unsigned short)(u >> 16);
}
// fast packed f32x2 -> bf16x2: +0x8000 round then byte-perm the two high halves
static __device__ __forceinline__ unsigned int pk2r(float lo, float hi) {
  unsigned int ul = __float_as_uint(lo) + 0x8000u;
  unsigned int uh = __float_as_uint(hi) + 0x8000u;
  return __builtin_amdgcn_perm(uh, ul, 0x07060302u);  // {uh[31:16], ul[31:16]}
}

// Load one pre-swizzled 16x32 bf16 A-fragment from LDS as raw dwords
// (fragment = 32 lanes x 16 halfs contiguous, 1 KB; two ds_load_b128 per lane)
static __device__ __forceinline__ v8u ld_frag_u(const unsigned short* base,
                                                int frag, int lane) {
  const uint4* p = (const uint4*)(base + frag * 512 + lane * 16);
  uint4 a = p[0];
  uint4 b = p[1];
  v8u u;
  u[0] = a.x; u[1] = a.y; u[2] = a.z; u[3] = a.w;
  u[4] = b.x; u[5] = b.y; u[6] = b.z; u[7] = b.w;
  return u;
}
#define BF(u) __builtin_bit_cast(v16bf, (u))

#define WMMA_BF16(A, B, C) \
  __builtin_amdgcn_wmma_f32_16x16x32_bf16(false, (A), false, (B), (short)0, (C), false, false)

__global__ __launch_bounds__(NTHREADS)
void Seq2Seq_71107478553198_kernel(const float* __restrict__ x,
                                   const float* __restrict__ enc_Wx,
                                   const float* __restrict__ enc_bx,
                                   const float* __restrict__ enc_Wh,
                                   const float* __restrict__ dec_Wx,
                                   const float* __restrict__ dec_bx,
                                   const float* __restrict__ dec_Wh,
                                   const float* __restrict__ dec_Wy,
                                   const float* __restrict__ dec_by,
                                   float* __restrict__ out) {
  // ---- LDS: 128 KB Wh A-fragments (enc then dec) + 24 KB aux = 152 KB ----
  __shared__ __attribute__((aligned(16))) unsigned short sh_fragW[128 * 512];
  __shared__ __attribute__((aligned(16))) unsigned short sh_aux[12288];

  const int tid    = threadIdx.x;
  const int lane   = tid & 31;
  const int waveId = tid >> 5;
  const int n0     = lane & 15;   // batch column within this wave's 16-row tile
  const int hi     = lane >> 4;   // half-wave select
  const int rowBase = blockIdx.x * ROWS_PER_BLOCK + waveId * ROWS_PER_WAVE;
  const int myRow   = rowBase + n0;

  // ====== setup: encoder weights as 16x32 A-fragments ======
  // D[d,n] = sum_k Wh[d,k] * h[n,k]  ->  A[m][k] = Wh[dt*16+m][kc*32+k]
  // A layout: lane<16: halfs 0..15 -> K0..15; lane>=16: +16
  for (int e = tid; e < 128 * 512; e += NTHREADS) {
    int f = e >> 9, slot = e & 511, ln = slot >> 4, h = slot & 15;
    int kc = f >> 4, dt = f & 15;
    int k = kc * 32 + ((ln >> 4) << 4) + h;
    int d = dt * 16 + (ln & 15);
    sh_fragW[e] = f2bf_rne(enc_Wh[d * DD + k]);
  }
  // enc Wx A-frags (16 d-tiles, single K-chunk). k<12: Wx; k==12: bias row; else 0.
  for (int e = tid; e < 16 * 512; e += NTHREADS) {
    int f = e >> 9, slot = e & 511, ln = slot >> 4, h = slot & 15;
    int k = ((ln >> 4) << 4) + h;
    int d = f * 16 + (ln & 15);
    float v = 0.0f;
    if (k < CC) v = enc_Wx[d * CC + k];
    else if (k == CC) v = enc_bx[d];
    sh_aux[e] = f2bf_rne(v);
  }
  __syncthreads();

  // h^T as 8 register-resident 32x16 B-fragments (B: lane(n,hi), half h -> k=16*hi+h)
  v8u hB[8];
#pragma unroll
  for (int kc = 0; kc < 8; ++kc)
#pragma unroll
    for (int i = 0; i < 8; ++i) hB[kc][i] = 0u;

  v8f zc;
#pragma unroll
  for (int i = 0; i < 8; ++i) zc[i] = 0.0f;

  // ====== encoder recurrence ======
  int fb0 = 0;   // laundered fragment base: defeats LICM of the weight ds_loads
  for (int t = 0; t < T_STEPS; ++t) {
    asm volatile("" : "+v"(fb0));

    // x^T B-fragment: k = 16*hi + h; k<12 from x row, k==12 -> 1.0 (bias row)
    v8u xu;
#pragma unroll
    for (int i = 0; i < 8; ++i) xu[i] = 0u;
    if (hi == 0) {
      const float* xr = x + ((size_t)t * NN + myRow) * CC;
      float4 f0 = *(const float4*)xr;
      float4 f1 = *(const float4*)(xr + 4);
      float4 f2 = *(const float4*)(xr + 8);
      xu[0] = pk2r(f0.x, f0.y); xu[1] = pk2r(f0.z, f0.w);
      xu[2] = pk2r(f1.x, f1.y); xu[3] = pk2r(f1.z, f1.w);
      xu[4] = pk2r(f2.x, f2.y); xu[5] = pk2r(f2.z, f2.w);
      xu[6] = 0x00003f80u;   // k=12 -> bf16(1.0)
    }
    v16bf xB = __builtin_bit_cast(v16bf, xu);

    // Ping-pong fragment buffers: issue Wx loads + Wh chunk-0 loads up front,
    // then keep one chunk of loads in flight behind each 16-WMMA group.
    v8u fb[2][16];
#pragma unroll
    for (int dt = 0; dt < 16; ++dt) fb[1][dt] = ld_frag_u(sh_aux, fb0 + dt, lane);      // Wx
#pragma unroll
    for (int dt = 0; dt < 16; ++dt) fb[0][dt] = ld_frag_u(sh_fragW, fb0 + dt, lane);    // Wh g0

    v8f acc[16];
#pragma unroll
    for (int dt = 0; dt < 16; ++dt)
      acc[dt] = WMMA_BF16(BF(fb[1][dt]), xB, zc);     // x-projection (+bias row), C = 0

#pragma unroll
    for (int kc = 0; kc < 8; ++kc) {
      const int cur = kc & 1, nxt = cur ^ 1;
      if (kc < 7) {
#pragma unroll
        for (int dt = 0; dt < 16; ++dt)
          fb[nxt][dt] = ld_frag_u(sh_fragW, fb0 + (kc + 1) * 16 + dt, lane);
      }
      v16bf b = BF(hB[kc]);
#pragma unroll
      for (int dt = 0; dt < 16; ++dt)
        acc[dt] = WMMA_BF16(BF(fb[cur][dt]), b, acc[dt]);
    }

    // relu + pack to bf16x2, then half-wave exchange to rebuild h^T B-fragments
    unsigned int pk[16][4];
#pragma unroll
    for (int ct = 0; ct < 16; ++ct)
#pragma unroll
      for (int j = 0; j < 4; ++j)
        pk[ct][j] = pk2r(fmaxf(acc[ct][2 * j], 0.0f), fmaxf(acc[ct][2 * j + 1], 0.0f));
#pragma unroll
    for (int kc = 0; kc < 8; ++kc) {
#pragma unroll
      for (int j = 0; j < 4; ++j) {
        unsigned int send = hi ? pk[2 * kc][j] : pk[2 * kc + 1][j];
        unsigned int recv = (unsigned int)__shfl_xor((int)send, 16, 32);
        hB[kc][j]     = hi ? recv : pk[2 * kc][j];
        hB[kc][4 + j] = hi ? pk[2 * kc + 1][j] : recv;
      }
    }
  }

  // ====== setup: decoder weights ======
  __syncthreads();
  for (int e = tid; e < 128 * 512; e += NTHREADS) {
    int f = e >> 9, slot = e & 511, ln = slot >> 4, h = slot & 15;
    int kc = f >> 4, dt = f & 15;
    int k = kc * 32 + ((ln >> 4) << 4) + h;
    int d = dt * 16 + (ln & 15);
    sh_fragW[e] = f2bf_rne(dec_Wh[d * DD + k]);
  }
  // Wy A-frags (8 K-chunks; output rows kk, rows 12..15 zero) at aux[0..4095]
  for (int e = tid; e < 8 * 512; e += NTHREADS) {
    int f = e >> 9, slot = e & 511, ln = slot >> 4, h = slot & 15;
    int k = f * 32 + ((ln >> 4) << 4) + h;
    int m = ln & 15;
    sh_aux[e] = (m < KK) ? f2bf_rne(dec_Wy[m * DD + k]) : (unsigned short)0;
  }
  // dec Wx A-frags (+bias row at k==12) at aux[4096..12287]
  for (int e = tid; e < 16 * 512; e += NTHREADS) {
    int f = e >> 9, slot = e & 511, ln = slot >> 4, h = slot & 15;
    int k = ((ln >> 4) << 4) + h;
    int d = f * 16 + (ln & 15);
    float v = 0.0f;
    if (k < CC) v = dec_Wx[d * CC + k];
    else if (k == CC) v = dec_bx[d];
    sh_aux[4096 + e] = f2bf_rne(v);
  }
  __syncthreads();

  // per-lane by bias vector for the y accumulator (kk = v + 8*hi)
  v8f byC;
#pragma unroll
  for (int v = 0; v < 8; ++v) {
    int kk = v + 8 * hi;
    byC[v] = (kk < KK) ? dec_by[kk] : 0.0f;
  }

  int rowIdx = CC - 2;   // sos0 one-hot index = 10, same for every row

  // ====== decoder greedy loop ======
  int fb1 = 0;   // laundered fragment base for decoder weight streams
  for (int t = 0; t < T_STEPS; ++t) {
    asm volatile("" : "+v"(fb1));

    // sos^T B-fragment: one-hot at rowIdx (c<12) + bias row 1.0 at c==12; hi half all zero
    v8u su;
#pragma unroll
    for (int i = 0; i < 8; ++i) su[i] = 0u;
    if (hi == 0) {
#pragma unroll
      for (int j = 0; j < 6; ++j)
        su[j] = ((rowIdx >> 1) == j) ? ((rowIdx & 1) ? 0x3f800000u : 0x00003f80u) : 0u;
      su[6] = 0x00003f80u;
    }
    v16bf sB = __builtin_bit_cast(v16bf, su);

    v8u fb[2][16];
#pragma unroll
    for (int dt = 0; dt < 16; ++dt) fb[1][dt] = ld_frag_u(sh_aux + 4096, fb1 + dt, lane); // dec Wx
#pragma unroll
    for (int dt = 0; dt < 16; ++dt) fb[0][dt] = ld_frag_u(sh_fragW, fb1 + dt, lane);      // Wh g0

    v8f acc[16];
#pragma unroll
    for (int dt = 0; dt < 16; ++dt)
      acc[dt] = WMMA_BF16(BF(fb[1][dt]), sB, zc);     // one-hot projection (+bias row)

#pragma unroll
    for (int kc = 0; kc < 8; ++kc) {
      const int cur = kc & 1, nxt = cur ^ 1;
      if (kc < 7) {
#pragma unroll
        for (int dt = 0; dt < 16; ++dt)
          fb[nxt][dt] = ld_frag_u(sh_fragW, fb1 + (kc + 1) * 16 + dt, lane);
      }
      v16bf b = BF(hB[kc]);
#pragma unroll
      for (int dt = 0; dt < 16; ++dt)
        acc[dt] = WMMA_BF16(BF(fb[cur][dt]), b, acc[dt]);
    }

    // relu + repack h^T
    unsigned int pk[16][4];
#pragma unroll
    for (int ct = 0; ct < 16; ++ct)
#pragma unroll
      for (int j = 0; j < 4; ++j)
        pk[ct][j] = pk2r(fmaxf(acc[ct][2 * j], 0.0f), fmaxf(acc[ct][2 * j + 1], 0.0f));
#pragma unroll
    for (int kc = 0; kc < 8; ++kc) {
#pragma unroll
      for (int j = 0; j < 4; ++j) {
        unsigned int send = hi ? pk[2 * kc][j] : pk[2 * kc + 1][j];
        unsigned int recv = (unsigned int)__shfl_xor((int)send, 16, 32);
        hB[kc][j]     = hi ? recv : pk[2 * kc][j];
        hB[kc][4 + j] = hi ? pk[2 * kc + 1][j] : recv;
      }
    }

    // y tile: D[kk, n] = Wy @ h_new^T + by  (Wy streamed: 8 frag loads then the chain)
    v8u fy[8];
#pragma unroll
    for (int kc = 0; kc < 8; ++kc) fy[kc] = ld_frag_u(sh_aux, fb1 + kc, lane);
    v8f y = WMMA_BF16(BF(fy[0]), BF(hB[0]), byC);
#pragma unroll
    for (int kc = 1; kc < 8; ++kc)
      y = WMMA_BF16(BF(fy[kc]), BF(hB[kc]), y);

    // store ys[t, myRow, :]: lane holds kk = v + 8*hi for batch row n0 -> coalesced b128
    size_t off = ((size_t)t * NN + myRow) * KK;
    if (hi == 0) {
      *(float4*)(out + off)     = make_float4(y[0], y[1], y[2], y[3]);
      *(float4*)(out + off + 4) = make_float4(y[4], y[5], y[6], y[7]);
    } else {
      *(float4*)(out + off + 8) = make_float4(y[0], y[1], y[2], y[3]);
    }

    // greedy argmax over 12 logits: local scan + one half-wave exchange
    float bv = y[0];
    int   bk = 8 * hi;
#pragma unroll
    for (int v = 1; v < 8; ++v) {
      int kk = v + 8 * hi;
      float cand = (kk < KK) ? y[v] : -3.0e38f;
      if (cand > bv) { bv = cand; bk = kk; }   // ascending order keeps first max
    }
    float pv  = __shfl_xor(bv, 16, 32);
    int   pkk = __shfl_xor(bk, 16, 32);
    if (pv > bv || (pv == bv && pkk < bk)) { bv = pv; bk = pkk; }
    rowIdx = bk;
  }
}

extern "C" void kernel_launch(void* const* d_in, const int* in_sizes, int n_in,
                              void* d_out, int out_size, void* d_ws, size_t ws_size,
                              hipStream_t stream) {
  (void)in_sizes; (void)n_in; (void)out_size; (void)d_ws; (void)ws_size;
  const float* x      = (const float*)d_in[0];
  const float* enc_Wx = (const float*)d_in[1];
  const float* enc_bx = (const float*)d_in[2];
  const float* enc_Wh = (const float*)d_in[3];
  // d_in[4] = enc_Wy, d_in[5] = enc_by: computed-and-discarded in the reference
  const float* dec_Wx = (const float*)d_in[6];
  const float* dec_bx = (const float*)d_in[7];
  const float* dec_Wh = (const float*)d_in[8];
  const float* dec_Wy = (const float*)d_in[9];
  const float* dec_by = (const float*)d_in[10];
  float* out = (float*)d_out;

  Seq2Seq_71107478553198_kernel<<<dim3(NBLOCKS), dim3(NTHREADS), 0, stream>>>(
      x, enc_Wx, enc_bx, enc_Wh, dec_Wx, dec_bx, dec_Wh, dec_Wy, dec_by, out);
}
